// RecurrentSlotAttention_58729382805766
// MI455X (gfx1250) — matrix operand order repaired and checked by reading, not compiled, not run on hardware
//
#include <hip/hip_runtime.h>
#include <math.h>

typedef __bf16 bf16_t;
typedef __attribute__((ext_vector_type(16))) __bf16 v16bf;
typedef __attribute__((ext_vector_type(8)))  float  v8f;

#define NSLOT   8
#define DSLOT   256
#define DFEAT   768
#define NTOK    4096
#define NBATCH  32
#define ATTN_SCALE 0.0625f   /* 256^-0.5 */
#define EPSV    1e-8f

union V16U { uint4 u[2]; v16bf v; };
union F8U  { bf16_t h[8]; uint4 u; };

// A-matrix 16x32 bf16 fragment (ISA 7.12.2 16-bit A layout):
// lane half 0: K = {0..7, 16..23}; half 1: K = {8..15, 24..31}
template <typename T>
static __device__ inline v16bf loadA_frag(const T* base, int ld, int k0, int lane) {
  int m = lane & 15, half = lane >> 4;
  const T* p = base + (size_t)m * ld + k0 + half * 8;
  V16U r;
  r.u[0] = *(const uint4*)(p);
  r.u[1] = *(const uint4*)(p + 16);
  return r.v;
}
// B-matrix 32x16 bf16 fragment: lane half 0 holds K={0..7,16..23}, half 1 K={8..15,24..31},
// column n = lane&15; source layout: element (k,n) at base[n*ld + k]
static __device__ inline v16bf loadB_frag(const bf16_t* base, int ld, int k0, int lane) {
  int c = lane & 15, half = lane >> 4;
  const bf16_t* p = base + (size_t)c * ld + k0 + half * 8;
  V16U r;
  r.u[0] = *(const uint4*)(p);
  r.u[1] = *(const uint4*)(p + 16);
  return r.v;
}
static __device__ inline v8f wmma_bf16(v16bf a, v16bf b, v8f c) {
  return __builtin_amdgcn_wmma_f32_16x16x32_bf16(false, a, false, b, (short)0, c, false, false);
}

// ---------------------------------------------------------------- weights->bf16
__global__ void k_convert_w(const float* __restrict__ Wk, const float* __restrict__ Wv,
                            bf16_t* __restrict__ Wk_b, bf16_t* __restrict__ Wv_b) {
  int i = blockIdx.x * blockDim.x + threadIdx.x;
  const int n = DSLOT * DFEAT;
  if (i < n) Wk_b[i] = (bf16_t)Wk[i];
  else { int j = i - n; if (j < n) Wv_b[j] = (bf16_t)Wv[j]; }
}

// ------------------------------------------------- LN(features) -> k, v (WMMA)
// One block per 16 rows. k stored (B*N, 256) bf16 row-major; v stored transposed
// (B, 256, 4096) bf16 so the updates-GEMM B fragments are contiguous.
__global__ void __launch_bounds__(256) k_ln_proj_kv(
    const float* __restrict__ feat, const float* __restrict__ g_in, const float* __restrict__ b_in,
    const bf16_t* __restrict__ Wk_b, const bf16_t* __restrict__ Wv_b,
    bf16_t* __restrict__ k_out, bf16_t* __restrict__ v_t) {
  __shared__ bf16_t fnS[16 * DFEAT];
  const int tid = threadIdx.x, lane = tid & 31, wave = tid >> 5;
  const int r0 = blockIdx.x * 16;  // global row = b*4096 + n
  // LayerNorm: each wave handles 2 rows of 768
  for (int rr = 0; rr < 2; ++rr) {
    int row = wave * 2 + rr;
    const float* src = feat + (size_t)(r0 + row) * DFEAT;
    float xv[24], s = 0.f, ss = 0.f;
#pragma unroll
    for (int j = 0; j < 24; ++j) { float x = src[lane + j * 32]; xv[j] = x; s += x; ss += x * x; }
#pragma unroll
    for (int off = 16; off > 0; off >>= 1) { s += __shfl_xor(s, off, 32); ss += __shfl_xor(ss, off, 32); }
    float mean = s * (1.f / DFEAT);
    float var  = ss * (1.f / DFEAT) - mean * mean;
    float rstd = rsqrtf(var + 1e-5f);
#pragma unroll
    for (int j = 0; j < 24; ++j) {
      int f = lane + j * 32;
      fnS[row * DFEAT + f] = (bf16_t)((xv[j] - mean) * rstd * g_in[f] + b_in[f]);
    }
  }
  __syncthreads();
  // GEMM: 32 output tiles (16 for k, 16 for v); 4 tiles per wave, K=768
  v8f acc[4] = {};
  const bf16_t* wbase[4]; int col0[4];
#pragma unroll
  for (int i = 0; i < 4; ++i) {
    int t = wave * 4 + i;
    wbase[i] = (t < 16) ? Wk_b : Wv_b;
    col0[i]  = (t & 15) * 16;
  }
  for (int kc = 0; kc < DFEAT / 32; ++kc) {
    v16bf a = loadA_frag(fnS, DFEAT, kc * 32, lane);
#pragma unroll
    for (int i = 0; i < 4; ++i) {
      v16bf b = loadB_frag(wbase[i] + (size_t)col0[i] * DFEAT, DFEAT, kc * 32, lane);
      acc[i] = wmma_bf16(a, b, acc[i]);
    }
  }
  // store (C layout: col = lane&15, rows m = j + 8*(lane>>4))
  int c = lane & 15, half = lane >> 4;
  int bidx = r0 / NTOK, nbase = r0 % NTOK;
#pragma unroll
  for (int i = 0; i < 4; ++i) {
    int t = wave * 4 + i;
    if (t < 16) {
      int h = col0[i] + c;
#pragma unroll
      for (int j = 0; j < 8; ++j)
        k_out[(size_t)(r0 + j + 8 * half) * DSLOT + h] = (bf16_t)acc[i][j];
    } else {
      int d = col0[i] + c;
      F8U pk;
#pragma unroll
      for (int j = 0; j < 8; ++j) pk.h[j] = (bf16_t)acc[i][j];
      *(uint4*)(v_t + ((size_t)bidx * DSLOT + d) * NTOK + nbase + 8 * half) = pk.u;
    }
  }
}

// ------------------------------------------------- LN(slots) -> q (bf16, 16x256 padded)
__global__ void __launch_bounds__(256) k_slot_q(
    const float* __restrict__ slots, const float* __restrict__ g_s, const float* __restrict__ b_s,
    const float* __restrict__ Wq, bf16_t* __restrict__ q_b, float* __restrict__ rowsum) {
  __shared__ float lnS[NSLOT * DSLOT];
  int b = blockIdx.x, tid = threadIdx.x, lane = tid & 31, wave = tid >> 5;
  const float* src = slots + ((size_t)b * NSLOT + wave) * DSLOT;
  float xv[8], s = 0.f, ss = 0.f;
#pragma unroll
  for (int j = 0; j < 8; ++j) { float x = src[lane + j * 32]; xv[j] = x; s += x; ss += x * x; }
#pragma unroll
  for (int off = 16; off > 0; off >>= 1) { s += __shfl_xor(s, off, 32); ss += __shfl_xor(ss, off, 32); }
  float mean = s * (1.f / DSLOT);
  float var  = ss * (1.f / DSLOT) - mean * mean;
  float rstd = rsqrtf(var + 1e-5f);
#pragma unroll
  for (int j = 0; j < 8; ++j) {
    int d = lane + j * 32;
    lnS[wave * DSLOT + d] = (xv[j] - mean) * rstd * g_s[d] + b_s[d];
  }
  __syncthreads();
  int h = tid;
  const float* wrow = Wq + (size_t)h * DSLOT;
  for (int k = 0; k < NSLOT; ++k) {
    float acc = 0.f;
    for (int d = 0; d < DSLOT; ++d) acc += lnS[k * DSLOT + d] * wrow[d];
    q_b[((size_t)b * 16 + k) * DSLOT + h]     = (bf16_t)acc;
    q_b[((size_t)b * 16 + 8 + k) * DSLOT + h] = (bf16_t)0.f;  // pad rows
  }
  if (tid < NSLOT) rowsum[b * NSLOT + tid] = 0.f;
}

// --------------------- logits = q k^T * scale, softmax over slot axis, rowsums
// grid = (8 n-chunks, 32 batches); each wave: 4 col tiles (64 columns)
__global__ void __launch_bounds__(256) k_logits(
    const bf16_t* __restrict__ q_b, const bf16_t* __restrict__ k_b,
    bf16_t* __restrict__ attn_b, float* __restrict__ attn_f32,
    float* __restrict__ rowsum) {
  int b = blockIdx.y, tid = threadIdx.x, lane = tid & 31, wave = tid >> 5;
  int ncol0 = blockIdx.x * 512 + wave * 64;
  const bf16_t* qbase = q_b + (size_t)b * 16 * DSLOT;
  v8f acc[4] = {};
  for (int kc = 0; kc < DSLOT / 32; ++kc) {
    v16bf a = loadA_frag(qbase, DSLOT, kc * 32, lane);
#pragma unroll
    for (int i = 0; i < 4; ++i) {
      const bf16_t* kb = k_b + ((size_t)b * NTOK + ncol0 + i * 16) * DSLOT;
      v16bf bb = loadB_frag(kb, DSLOT, kc * 32, lane);
      acc[i] = wmma_bf16(a, bb, acc[i]);
    }
  }
  float rs[8];
#pragma unroll
  for (int j = 0; j < 8; ++j) rs[j] = 0.f;
  if (lane < 16) {  // lanes 0-15 hold slot rows 0-7 of their column: lane-local softmax
#pragma unroll
    for (int i = 0; i < 4; ++i) {
      int n = ncol0 + i * 16 + lane;
      float l[8], mx = -1e30f;
#pragma unroll
      for (int j = 0; j < 8; ++j) { l[j] = acc[i][j] * ATTN_SCALE; mx = fmaxf(mx, l[j]); }
      float sum = 0.f;
#pragma unroll
      for (int j = 0; j < 8; ++j) { l[j] = __expf(l[j] - mx); sum += l[j]; }
      float inv = 1.f / sum;
#pragma unroll
      for (int j = 0; j < 8; ++j) {
        float a_ = l[j] * inv;
        rs[j] += a_;
        if (attn_f32) attn_f32[((size_t)b * NSLOT + j) * NTOK + n] = a_;
        else          attn_b [((size_t)b * 16    + j) * NTOK + n] = (bf16_t)a_;
      }
    }
  }
#pragma unroll
  for (int j = 0; j < 8; ++j) {
    float v = rs[j];
#pragma unroll
    for (int off = 16; off > 0; off >>= 1) v += __shfl_xor(v, off, 32);
    if (lane == 0) atomicAdd(&rowsum[b * NSLOT + j], v);
  }
}

// ------- updates = attn_n @ v (WMMA, K=4096), then GRU + LN + MLP -> new slots
__global__ void __launch_bounds__(256) k_update(
    const bf16_t* __restrict__ attn_b, const bf16_t* __restrict__ v_t,
    const float* __restrict__ rowsum,
    const float* __restrict__ W_ih, const float* __restrict__ W_hh,
    const float* __restrict__ b_ih, const float* __restrict__ b_hh,
    const float* __restrict__ g_m, const float* __restrict__ b_m,
    const float* __restrict__ W1, const float* __restrict__ b1,
    const float* __restrict__ W2, const float* __restrict__ b2,
    float* __restrict__ slots) {
  __shared__ float slL[NSLOT * DSLOT], updL[NSLOT * DSLOT], hL[NSLOT * DSLOT];
  __shared__ float lnhL[NSLOT * DSLOT], m1L[NSLOT * DSLOT];
  int b = blockIdx.x, tid = threadIdx.x, lane = tid & 31, wave = tid >> 5;
  for (int i = tid; i < NSLOT * DSLOT; i += 256) slL[i] = slots[(size_t)b * NSLOT * DSLOT + i];
  // WMMA: M=16(pad) x N=256 x K=4096; 2 col tiles per wave
  const bf16_t* abase = attn_b + (size_t)b * 16 * NTOK;
  v8f acc[2] = {};
  for (int kc = 0; kc < NTOK / 32; ++kc) {
    v16bf a = loadA_frag(abase, NTOK, kc * 32, lane);
#pragma unroll
    for (int i = 0; i < 2; ++i) {
      const bf16_t* vb = v_t + ((size_t)b * DSLOT + (wave * 2 + i) * 16) * NTOK;
      v16bf bb = loadB_frag(vb, NTOK, kc * 32, lane);
      acc[i] = wmma_bf16(a, bb, acc[i]);
    }
  }
  if (lane < 16) {  // fold 1/(rowsum+eps) row scale, write rows 0-7
    float inv[8];
#pragma unroll
    for (int j = 0; j < 8; ++j) inv[j] = 1.f / (rowsum[b * NSLOT + j] + EPSV);
#pragma unroll
    for (int i = 0; i < 2; ++i) {
      int d = (wave * 2 + i) * 16 + lane;
#pragma unroll
      for (int j = 0; j < 8; ++j) updL[j * DSLOT + d] = acc[i][j] * inv[j];
    }
  }
  __syncthreads();
  // GRU (torch gate order r,z,n): 2048 outputs over 256 threads
  for (int idx = tid; idx < NSLOT * DSLOT; idx += 256) {
    int k = idx >> 8, d = idx & 255;
    float gir = b_ih[d], giz = b_ih[256 + d], gin = b_ih[512 + d];
    float ghr = b_hh[d], ghz = b_hh[256 + d], ghn = b_hh[512 + d];
    const float* wr = W_ih + (size_t)d * DSLOT;
    const float* wz = W_ih + (size_t)(256 + d) * DSLOT;
    const float* wn = W_ih + (size_t)(512 + d) * DSLOT;
    const float* hr = W_hh + (size_t)d * DSLOT;
    const float* hz = W_hh + (size_t)(256 + d) * DSLOT;
    const float* hn = W_hh + (size_t)(512 + d) * DSLOT;
    for (int dd = 0; dd < DSLOT; ++dd) {
      float u = updL[k * DSLOT + dd], sl = slL[k * DSLOT + dd];
      gir += u * wr[dd]; giz += u * wz[dd]; gin += u * wn[dd];
      ghr += sl * hr[dd]; ghz += sl * hz[dd]; ghn += sl * hn[dd];
    }
    float r = 1.f / (1.f + __expf(-(gir + ghr)));
    float z = 1.f / (1.f + __expf(-(giz + ghz)));
    float nn = tanhf(gin + r * ghn);
    hL[idx] = (1.f - z) * nn + z * slL[idx];
  }
  __syncthreads();
  // LayerNorm(h): wave w owns slot row w
  {
    const float* src = hL + wave * DSLOT;
    float xv[8], s = 0.f, ss = 0.f;
#pragma unroll
    for (int j = 0; j < 8; ++j) { float x = src[lane + j * 32]; xv[j] = x; s += x; ss += x * x; }
#pragma unroll
    for (int off = 16; off > 0; off >>= 1) { s += __shfl_xor(s, off, 32); ss += __shfl_xor(ss, off, 32); }
    float mean = s * (1.f / DSLOT);
    float var  = ss * (1.f / DSLOT) - mean * mean;
    float rstd = rsqrtf(var + 1e-5f);
#pragma unroll
    for (int j = 0; j < 8; ++j) {
      int d = lane + j * 32;
      lnhL[wave * DSLOT + d] = (xv[j] - mean) * rstd * g_m[d] + b_m[d];
    }
  }
  __syncthreads();
  for (int idx = tid; idx < NSLOT * DSLOT; idx += 256) {
    int k = idx >> 8, m = idx & 255;
    float a = b1[m];
    const float* w = W1 + (size_t)m * DSLOT;
    for (int d = 0; d < DSLOT; ++d) a += lnhL[k * DSLOT + d] * w[d];
    m1L[idx] = fmaxf(a, 0.f);
  }
  __syncthreads();
  for (int idx = tid; idx < NSLOT * DSLOT; idx += 256) {
    int k = idx >> 8, d = idx & 255;
    float a = b2[d];
    const float* w = W2 + (size_t)d * DSLOT;
    for (int m = 0; m < DSLOT; ++m) a += m1L[k * DSLOT + m] * w[m];
    slots[(size_t)b * NSLOT * DSLOT + idx] = hL[idx] + a;
  }
}

// ----------------------------------- confidence blend: slots*mask + prev*(1-mask)
__global__ void __launch_bounds__(256) k_finalize(
    const float* __restrict__ slots, const float* __restrict__ prev,
    const float* __restrict__ rowsum, float* __restrict__ out_slots) {
  int b = blockIdx.x, tid = threadIdx.x;
  for (int idx = tid; idx < NSLOT * DSLOT; idx += 256) {
    int k = idx >> 8;
    float mean = rowsum[b * NSLOT + k] * (1.f / NTOK);
    float mask = 1.f / (1.f + __expf(-mean));
    size_t o = (size_t)b * NSLOT * DSLOT + idx;
    out_slots[o] = slots[o] * mask + prev[o] * (1.f - mask);
  }
}

extern "C" void kernel_launch(void* const* d_in, const int* in_sizes, int n_in,
                              void* d_out, int out_size, void* d_ws, size_t ws_size,
                              hipStream_t stream) {
  const float* feat = (const float*)d_in[0];
  const float* prev = (const float*)d_in[1];
  const float* g_in = (const float*)d_in[2];
  const float* b_in = (const float*)d_in[3];
  const float* g_s  = (const float*)d_in[4];
  const float* b_s  = (const float*)d_in[5];
  const float* g_m  = (const float*)d_in[6];
  const float* b_m  = (const float*)d_in[7];
  const float* Wq   = (const float*)d_in[8];
  const float* Wk   = (const float*)d_in[9];
  const float* Wv   = (const float*)d_in[10];
  const float* W_ih = (const float*)d_in[11];
  const float* W_hh = (const float*)d_in[12];
  const float* b_ih = (const float*)d_in[13];
  const float* b_hh = (const float*)d_in[14];
  const float* W1   = (const float*)d_in[15];
  const float* b1   = (const float*)d_in[16];
  const float* W2   = (const float*)d_in[17];
  const float* b2   = (const float*)d_in[18];

  char* ws = (char*)d_ws;
  size_t off = 0;
  auto carve = [&](size_t bytes) { void* p = ws + off; off = (off + bytes + 255) & ~(size_t)255; return p; };
  bf16_t* Wk_b   = (bf16_t*)carve((size_t)DSLOT * DFEAT * 2);
  bf16_t* Wv_b   = (bf16_t*)carve((size_t)DSLOT * DFEAT * 2);
  bf16_t* k_b    = (bf16_t*)carve((size_t)NBATCH * NTOK * DSLOT * 2);
  bf16_t* v_t    = (bf16_t*)carve((size_t)NBATCH * DSLOT * NTOK * 2);
  bf16_t* attn_b = (bf16_t*)carve((size_t)NBATCH * 16 * NTOK * 2);
  bf16_t* q_b    = (bf16_t*)carve((size_t)NBATCH * 16 * DSLOT * 2);
  float*  slots  = (float*)carve((size_t)NBATCH * NSLOT * DSLOT * 4);
  float*  rowsum = (float*)carve((size_t)NBATCH * NSLOT * 4);

  float* out_slots = (float*)d_out;
  float* out_attn  = (float*)d_out + (size_t)NBATCH * NSLOT * DSLOT;

  hipMemcpyAsync(slots, prev, (size_t)NBATCH * NSLOT * DSLOT * 4,
                 hipMemcpyDeviceToDevice, stream);
  k_convert_w<<<(2 * DSLOT * DFEAT) / 256, 256, 0, stream>>>(Wk, Wv, Wk_b, Wv_b);
  k_ln_proj_kv<<<(NBATCH * NTOK) / 16, 256, 0, stream>>>(feat, g_in, b_in, Wk_b, Wv_b, k_b, v_t);

  for (int it = 0; it < 3; ++it) {
    k_slot_q<<<NBATCH, 256, 0, stream>>>(slots, g_s, b_s, Wq, q_b, rowsum);
    k_logits<<<dim3(8, NBATCH), 256, 0, stream>>>(q_b, k_b, attn_b, nullptr, rowsum);
    k_update<<<NBATCH, 256, 0, stream>>>(attn_b, v_t, rowsum, W_ih, W_hh, b_ih, b_hh,
                                         g_m, b_m, W1, b1, W2, b2, slots);
  }
  // final attention pass: write fp32 attn to d_out, rowsums feed the mask
  k_slot_q<<<NBATCH, 256, 0, stream>>>(slots, g_s, b_s, Wq, q_b, rowsum);
  k_logits<<<dim3(8, NBATCH), 256, 0, stream>>>(q_b, k_b, attn_b, out_attn, rowsum);
  k_finalize<<<NBATCH, 256, 0, stream>>>(slots, prev, rowsum, out_slots);
}
